// XLMultiHeadsAttention_53085795779502
// MI455X (gfx1250) — matrix operand order repaired and checked by reading, compiled, not run on hardware
//
#include <hip/hip_runtime.h>
#include <hip/hip_bf16.h>
#include <cstddef>

typedef __attribute__((ext_vector_type(16))) _Float16 v16h;
typedef __attribute__((ext_vector_type(8)))  _Float16 v8h;
typedef __attribute__((ext_vector_type(8)))  float    v8f;

namespace {
constexpr int B_  = 2;
constexpr int S_  = 2048;
constexpr int M_  = 2048;
constexpr int E_  = 1024;
constexpr int H_  = 8;
constexpr int DH_ = 64;
constexpr int INNER_ = H_ * DH_;   // 512
constexpr int J_  = S_ + M_;       // 4096
constexpr float SCALE_ = 0.125f;   // DH^-0.5
}

#define WMMA16(A, Bm, C) \
  __builtin_amdgcn_wmma_f32_16x16x32_f16(false, (A), false, (Bm), (short)0, (C), false, false)

#if __has_builtin(__builtin_amdgcn_sched_barrier)
#define SCHED_FENCE() __builtin_amdgcn_sched_barrier(0)
#else
#define SCHED_FENCE()
#endif

// B-operand (32x16 f16): lane = column N (lane&15); 16 consecutive K values at
// K = 16*half + e  ->  one 32-byte contiguous per-lane load.
static __device__ __forceinline__ v16h ld16(const _Float16* p) {
  v8h a = *(const v8h*)p;
  v8h b = *(const v8h*)(p + 8);
  return __builtin_shufflevector(a, b, 0, 1, 2, 3, 4, 5, 6, 7,
                                 8, 9, 10, 11, 12, 13, 14, 15);
}

// A-operand (16x32 f16): lane = row M (lane&15); per-lane K
// [16*half .. 16*half+15] (same order as B) ->
// two contiguous 16-byte loads from a row pointer.
static __device__ __forceinline__ v16h ldA(const _Float16* row, int half) {
  v8h a = *(const v8h*)(row + 16 * half);
  v8h b = *(const v8h*)(row + 16 * half + 8);
  return __builtin_shufflevector(a, b, 0, 1, 2, 3, 4, 5, 6, 7,
                                 8, 9, 10, 11, 12, 13, 14, 15);
}

// ---------------------------------------------------------------------------
// Prep 1: X f32 [4096,1024] -> f16 (same layout)
// ---------------------------------------------------------------------------
__global__ __launch_bounds__(256) void xcvt_kernel(const float* __restrict__ in,
                                                   _Float16* __restrict__ out,
                                                   size_t total) {
  size_t i = (size_t)blockIdx.x * blockDim.x + threadIdx.x;
  if (i < total) out[i] = (_Float16)in[i];
}

// ---------------------------------------------------------------------------
// Prep 2: weight convert + transpose: in [K,N] f32 -> out [N,K] f16
// ---------------------------------------------------------------------------
__global__ __launch_bounds__(256) void wtcvt_kernel(const float* __restrict__ in,
                                                    _Float16* __restrict__ out,
                                                    int K, int N) {
  size_t i = (size_t)blockIdx.x * blockDim.x + threadIdx.x;
  if (i >= (size_t)K * N) return;
  int k = (int)(i / N);
  int n = (int)(i % N);
  out[(size_t)n * K + k] = (_Float16)in[i];
}

// ---------------------------------------------------------------------------
// Prep 3: xl_memory (f32 [B,M,2,INNER]) -> kh [B,H,J,DH] f16 (rows 0..M-1)
//                                          vh [B,H,DH,J] f16 (cols 0..M-1)
// ---------------------------------------------------------------------------
__global__ __launch_bounds__(256) void memcvt_kernel(const float* __restrict__ xl,
                                                     _Float16* __restrict__ kh,
                                                     _Float16* __restrict__ vh) {
  size_t idx = (size_t)blockIdx.x * blockDim.x + threadIdx.x;
  const size_t total = (size_t)B_ * M_ * INNER_;
  if (idx >= total) return;
  int f = (int)(idx % INNER_);
  size_t t = idx / INNER_;
  int s = (int)(t % M_);
  int b = (int)(t / M_);
  int h  = f >> 6;
  int dh = f & 63;
  const size_t base = ((size_t)(b * M_ + s) * 2) * INNER_ + f;
  kh[(((size_t)b * H_ + h) * J_ + s) * DH_ + dh] = (_Float16)xl[base];
  vh[(((size_t)b * H_ + h) * DH_ + dh) * J_ + s] = (_Float16)xl[base + INNER_];
}

// ---------------------------------------------------------------------------
// Kernel: QKV projection, one wave -> 16x64 output tile (4 WMMAs / K-step).
// Software-pipelined: iteration k+1's 10 b128 loads are issued before
// iteration k's WMMA burst; sched_barrier fences keep the order so each WMMA
// waits on loads issued a full iteration earlier.
// ---------------------------------------------------------------------------
__global__ __launch_bounds__(256) void qkv_gemm_kernel(
    const _Float16* __restrict__ xh,
    const _Float16* __restrict__ wqh, const float* __restrict__ bq,
    const _Float16* __restrict__ wkh, const float* __restrict__ bk,
    const _Float16* __restrict__ wvh, const float* __restrict__ bv,
    _Float16* __restrict__ qh, _Float16* __restrict__ kh, _Float16* __restrict__ vh,
    float* __restrict__ kvout) {
  const int ROWT = (B_ * S_) / 16;   // 256
  const int COLG = INNER_ / 64;      // 8 column-groups of 64
  const int TPW  = ROWT * COLG;      // 2048 per weight
  int wave = (int)((blockIdx.x * blockDim.x + threadIdx.x) >> 5);
  int lane = threadIdx.x & 31;
  int w = wave / TPW;
  if (w >= 3) return;
  int rem = wave % TPW;
  int rt = rem / COLG;
  int cg = rem % COLG;
  const _Float16* Wt = (w == 0) ? wqh : (w == 1) ? wkh : wvh;
  const float*    bp = (w == 0) ? bq  : (w == 1) ? bk  : bv;

  int half = lane >> 4;
  int lcol = lane & 15;
  int arow = rt * 16 + lcol;

  v8f acc[4] = {v8f{}, v8f{}, v8f{}, v8f{}};
  const _Float16* xrow = xh + (size_t)arow * E_;
  const _Float16* wb   = Wt + (size_t)(cg * 64 + lcol) * E_ + 16 * half;

  v16h a_cur = ldA(xrow, half);
  v16h bcur[4];
#pragma unroll
  for (int nt = 0; nt < 4; ++nt) bcur[nt] = ld16(wb + (size_t)(nt * 16) * E_);

#pragma unroll 2
  for (int kk = 32; kk < E_; kk += 32) {
    v16h a_nxt = ldA(xrow + kk, half);
    v16h bnxt[4];
#pragma unroll
    for (int nt = 0; nt < 4; ++nt)
      bnxt[nt] = ld16(wb + (size_t)(nt * 16) * E_ + kk);
    SCHED_FENCE();
#pragma unroll
    for (int nt = 0; nt < 4; ++nt)
      acc[nt] = WMMA16(a_cur, bcur[nt], acc[nt]);
    SCHED_FENCE();
    a_cur = a_nxt;
#pragma unroll
    for (int nt = 0; nt < 4; ++nt) bcur[nt] = bnxt[nt];
  }
#pragma unroll
  for (int nt = 0; nt < 4; ++nt)
    acc[nt] = WMMA16(a_cur, bcur[nt], acc[nt]);

#pragma unroll
  for (int nt = 0; nt < 4; ++nt) {
    int gcol = cg * 64 + nt * 16 + lcol;
    float bias = bp[gcol];
    int h  = gcol >> 6;
    int dh = gcol & 63;
#pragma unroll
    for (int r = 0; r < 8; ++r) {
      int row = rt * 16 + r + 8 * half;   // D-layout row
      int bb = row >> 11;
      int ss = row & (S_ - 1);
      float val = acc[nt][r] + bias;
      if (w == 0) {
        qh[(((size_t)bb * H_ + h) * S_ + ss) * DH_ + dh] = (_Float16)val;
      } else if (w == 1) {
        kh[(((size_t)bb * H_ + h) * J_ + (M_ + ss)) * DH_ + dh] = (_Float16)val;
        kvout[(((size_t)bb * S_ + ss) * 2 + 0) * INNER_ + gcol] = val;
      } else {
        vh[(((size_t)bb * H_ + h) * DH_ + dh) * J_ + (M_ + ss)] = (_Float16)val;
        kvout[(((size_t)bb * S_ + ss) * 2 + 1) * INNER_ + gcol] = val;
      }
    }
  }
}

// ---------------------------------------------------------------------------
// Kernel: flash attention. One wave per 16-query tile per (b,h).
// V-tile loads are issued right after the QK^T WMMAs so they are serviced
// while the online-softmax VALU block runs.
// ---------------------------------------------------------------------------
__global__ __launch_bounds__(256) void attn_kernel(
    const _Float16* __restrict__ qh, const _Float16* __restrict__ kh,
    const _Float16* __restrict__ vh, const float* __restrict__ rel,
    _Float16* __restrict__ oh) {
  __shared__ _Float16 pbuf[8][16 * 32];   // per-wave P staging (D- -> A-layout)
  int wib  = threadIdx.x >> 5;
  int lane = threadIdx.x & 31;
  int wave = (int)(blockIdx.x * (blockDim.x >> 5) + wib);
  const int QT = S_ / 16;   // 128
  int qt = wave % QT;
  int bh = wave / QT;
  int h = bh % H_;
  int b = bh / H_;
  int half = lane >> 4;
  int lcol = lane & 15;

  const _Float16* qbase = qh + (((size_t)b * H_ + h) * S_) * DH_;
  const _Float16* kbase = kh + (((size_t)b * H_ + h) * J_) * DH_;
  const _Float16* vbase = vh + (((size_t)b * H_ + h) * DH_) * J_;
  const float*  relbase = rel + ((size_t)h * S_) * J_;
  _Float16* pls = &pbuf[wib][0];

  const _Float16* qrow = qbase + (size_t)(qt * 16 + lcol) * DH_;
  v16h qa0 = ldA(qrow, half);        // dh 0..31
  v16h qa1 = ldA(qrow + 32, half);   // dh 32..63

  float mstat[8], lstat[8];
  v8f oacc[4] = {v8f{}, v8f{}, v8f{}, v8f{}};
#pragma unroll
  for (int r = 0; r < 8; ++r) { mstat[r] = -1e30f; lstat[r] = 0.0f; }

  const int jend = qt * 16 + 16 + M_;   // causal clip: keys j <= i + M
  const _Float16* vrow = vbase + (size_t)lcol * J_ + 16 * half;

  for (int j0 = 0; j0 < jend; j0 += 32) {
    // prefetch next chunk's rel_pos rows (dominant HBM stream)
    if (j0 + 32 < jend)
      __builtin_prefetch((const void*)(relbase + (size_t)(qt * 16 + lcol) * J_ +
                                       j0 + 32 + 16 * half), 0, 1);
    // --- preload 4 K B-operands, then S = Q K^T (two 16-key subtiles) ---
    const _Float16* kr0 = kbase + (size_t)(j0 + lcol) * DH_ + 16 * half;
    const _Float16* kr1 = kr0 + 16 * DH_;
    v16h kbm[4];
    kbm[0] = ld16(kr0);
    kbm[1] = ld16(kr0 + 32);
    kbm[2] = ld16(kr1);
    kbm[3] = ld16(kr1 + 32);
    v8f s0 = {}, s1 = {};
    s0 = WMMA16(qa0, kbm[0], s0);
    s0 = WMMA16(qa1, kbm[1], s0);
    s1 = WMMA16(qa0, kbm[2], s1);
    s1 = WMMA16(qa1, kbm[3], s1);
    // --- issue V loads now; they complete during the softmax VALU block ---
    v16h vbm[4];
#pragma unroll
    for (int nt = 0; nt < 4; ++nt)
      vbm[nt] = ld16(vrow + (size_t)(nt * 16) * J_ + j0);
    SCHED_FENCE();
    // --- bias + scale + mask + online softmax (rows = r + 8*half) ---
#pragma unroll
    for (int r = 0; r < 8; ++r) {
      int m  = r + 8 * half;
      int ig = qt * 16 + m;
      int jg0 = j0 + lcol;
      int jg1 = jg0 + 16;
      const float* rr = relbase + (size_t)ig * J_;
      float v0 = (s0[r] + rr[jg0]) * SCALE_;
      float v1 = (s1[r] + rr[jg1]) * SCALE_;
      if (jg0 > ig + M_) v0 = -1e30f;
      if (jg1 > ig + M_) v1 = -1e30f;
      float rmax = fmaxf(v0, v1);
#pragma unroll
      for (int off = 1; off < 16; off <<= 1)
        rmax = fmaxf(rmax, __shfl_xor(rmax, off, 32));
      float nm = fmaxf(mstat[r], rmax);
      float alpha = __expf(mstat[r] - nm);
      float p0 = __expf(v0 - nm);
      float p1 = __expf(v1 - nm);
      float ps = p0 + p1;
#pragma unroll
      for (int off = 1; off < 16; off <<= 1)
        ps += __shfl_xor(ps, off, 32);
      lstat[r] = lstat[r] * alpha + ps;
      mstat[r] = nm;
      oacc[0][r] *= alpha;
      oacc[1][r] *= alpha;
      oacc[2][r] *= alpha;
      oacc[3][r] *= alpha;
      pls[m * 32 + lcol]      = (_Float16)p0;
      pls[m * 32 + 16 + lcol] = (_Float16)p1;
    }
    // --- P (A-layout from LDS) x V ---
    v16h pa = ldA(pls + lcol * 32, half);
#pragma unroll
    for (int nt = 0; nt < 4; ++nt)
      oacc[nt] = WMMA16(pa, vbm[nt], oacc[nt]);
  }

  // --- normalize and store merged-head O as f16 [B, S, H*DH] ---
#pragma unroll
  for (int r = 0; r < 8; ++r) {
    int m = r + 8 * half;
    int srow = qt * 16 + m;
    float inv = 1.0f / lstat[r];
    _Float16* orow = oh + (((size_t)b * S_ + srow) * H_ + h) * DH_;
#pragma unroll
    for (int nt = 0; nt < 4; ++nt)
      orow[nt * 16 + lcol] = (_Float16)(oacc[nt][r] * inv);
  }
}

// ---------------------------------------------------------------------------
// Kernel: output projection, one wave -> 16x64 tile of out = O @ Wo + bo.
// Same software pipelining as the QKV GEMM.
// ---------------------------------------------------------------------------
__global__ __launch_bounds__(256) void oproj_kernel(const _Float16* __restrict__ oh,
                                                    const _Float16* __restrict__ woh,
                                                    const float* __restrict__ bo,
                                                    float* __restrict__ out) {
  const int COLG = E_ / 64;   // 16
  int wave = (int)((blockIdx.x * blockDim.x + threadIdx.x) >> 5);
  int lane = threadIdx.x & 31;
  int rt = wave / COLG;
  int cg = wave % COLG;
  if (rt >= (B_ * S_) / 16) return;
  int half = lane >> 4;
  int lcol = lane & 15;
  int arow = rt * 16 + lcol;

  v8f acc[4] = {v8f{}, v8f{}, v8f{}, v8f{}};
  const _Float16* orow = oh + (size_t)arow * INNER_;
  const _Float16* wb   = woh + (size_t)(cg * 64 + lcol) * INNER_ + 16 * half;

  v16h a_cur = ldA(orow, half);
  v16h bcur[4];
#pragma unroll
  for (int nt = 0; nt < 4; ++nt) bcur[nt] = ld16(wb + (size_t)(nt * 16) * INNER_);

#pragma unroll 2
  for (int kk = 32; kk < INNER_; kk += 32) {
    v16h a_nxt = ldA(orow + kk, half);
    v16h bnxt[4];
#pragma unroll
    for (int nt = 0; nt < 4; ++nt)
      bnxt[nt] = ld16(wb + (size_t)(nt * 16) * INNER_ + kk);
    SCHED_FENCE();
#pragma unroll
    for (int nt = 0; nt < 4; ++nt)
      acc[nt] = WMMA16(a_cur, bcur[nt], acc[nt]);
    SCHED_FENCE();
    a_cur = a_nxt;
#pragma unroll
    for (int nt = 0; nt < 4; ++nt) bcur[nt] = bnxt[nt];
  }
#pragma unroll
  for (int nt = 0; nt < 4; ++nt)
    acc[nt] = WMMA16(a_cur, bcur[nt], acc[nt]);

#pragma unroll
  for (int nt = 0; nt < 4; ++nt) {
    int gcol = cg * 64 + nt * 16 + lcol;
    float bias = bo[gcol];
#pragma unroll
    for (int r = 0; r < 8; ++r) {
      int row = rt * 16 + r + 8 * half;
      out[(size_t)row * E_ + gcol] = acc[nt][r] + bias;
    }
  }
}

// ---------------------------------------------------------------------------
extern "C" void kernel_launch(void* const* d_in, const int* in_sizes, int n_in,
                              void* d_out, int out_size, void* d_ws, size_t ws_size,
                              hipStream_t stream) {
  (void)in_sizes; (void)n_in; (void)out_size; (void)ws_size;
  const float* x   = (const float*)d_in[0];
  const float* rel = (const float*)d_in[1];
  const float* xl  = (const float*)d_in[2];
  const float* Wq  = (const float*)d_in[3];
  const float* bq  = (const float*)d_in[4];
  const float* Wk  = (const float*)d_in[5];
  const float* bk  = (const float*)d_in[6];
  const float* Wv  = (const float*)d_in[7];
  const float* bv  = (const float*)d_in[8];
  const float* Wo  = (const float*)d_in[9];
  const float* bo  = (const float*)d_in[10];

  float* out   = (float*)d_out;                       // [B,S,E]
  float* kvout = out + (size_t)B_ * S_ * E_;          // [B,S,2,INNER]

  // workspace carve-up (f16)
  _Float16* xh  = (_Float16*)d_ws;                       // [B*S, E]       8MB
  _Float16* qh  = xh  + (size_t)B_ * S_ * E_;            // [B,H,S,DH]     4MB
  _Float16* kh  = qh  + (size_t)B_ * H_ * S_ * DH_;      // [B,H,J,DH]     8MB
  _Float16* vh  = kh  + (size_t)B_ * H_ * J_ * DH_;      // [B,H,DH,J]     8MB
  _Float16* oh  = vh  + (size_t)B_ * H_ * J_ * DH_;      // [B*S, INNER]   4MB
  _Float16* wqh = oh  + (size_t)B_ * S_ * INNER_;        // [INNER, E]     1MB
  _Float16* wkh = wqh + (size_t)INNER_ * E_;
  _Float16* wvh = wkh + (size_t)INNER_ * E_;
  _Float16* woh = wvh + (size_t)INNER_ * E_;             // [E, INNER]     1MB

  // 1) precision/layout prep
  {
    const size_t tx = (size_t)B_ * S_ * E_;
    xcvt_kernel<<<dim3((unsigned)((tx + 255) / 256)), 256, 0, stream>>>(x, xh, tx);
    const unsigned gw = (unsigned)(((size_t)E_ * INNER_ + 255) / 256);
    wtcvt_kernel<<<dim3(gw), 256, 0, stream>>>(Wq, wqh, E_, INNER_);
    wtcvt_kernel<<<dim3(gw), 256, 0, stream>>>(Wk, wkh, E_, INNER_);
    wtcvt_kernel<<<dim3(gw), 256, 0, stream>>>(Wv, wvh, E_, INNER_);
    wtcvt_kernel<<<dim3(gw), 256, 0, stream>>>(Wo, woh, INNER_, E_);
    const size_t tm = (size_t)B_ * M_ * INNER_;
    memcvt_kernel<<<dim3((unsigned)((tm + 255) / 256)), 256, 0, stream>>>(xl, kh, vh);
  }
  // 2) QKV projection (also writes current-KV output block)
  {
    const int waves = 3 * ((B_ * S_) / 16) * (INNER_ / 64);   // 6144
    qkv_gemm_kernel<<<dim3((unsigned)(waves * 32 / 256)), 256, 0, stream>>>(
        xh, wqh, bq, wkh, bk, wvh, bv, qh, kh, vh, kvout);
  }
  // 3) flash attention
  {
    const int waves = B_ * H_ * (S_ / 16);   // 2048
    attn_kernel<<<dim3((unsigned)(waves * 32 / 256)), 256, 0, stream>>>(
        qh, kh, vh, rel, oh);
  }
  // 4) output projection
  {
    const int waves = ((B_ * S_) / 16) * (E_ / 64);   // 4096
    oproj_kernel<<<dim3((unsigned)(waves * 32 / 256)), 256, 0, stream>>>(
        oh, woh, bo, out);
  }
}
